// TDM_7284264534538
// MI455X (gfx1250) — matrix operand order repaired and checked by reading, not compiled, not run on hardware
//
#include <hip/hip_runtime.h>
#include <math.h>

// Problem constants (from reference)
#define B_   4
#define C_   64
#define RC_  192
#define H_   128
#define W_   128
#define HW_  (H_*W_)

typedef __attribute__((ext_vector_type(16))) __bf16 v16bf;
typedef __attribute__((ext_vector_type(8)))  __bf16 v8bf;
typedef __attribute__((ext_vector_type(8)))  float  v8f;

union AFrag { v16bf v; v8bf h[2]; };

__device__ __forceinline__ unsigned short f2bfbits(float f) {
    union { float f; unsigned u; } a; a.f = f;
    unsigned r = (a.u + 0x7FFFu + ((a.u >> 16) & 1u)) >> 16;  // RNE
    return (unsigned short)r;
}
__device__ __forceinline__ __bf16 bits2bf(unsigned short u) {
    union { unsigned short u; __bf16 h; } c; c.u = u; return c.h;
}
__device__ __forceinline__ __bf16 f2bf(float f) { return bits2bf(f2bfbits(f)); }
__device__ __forceinline__ int iclamp(int v, int lo, int hi) {
    return v < lo ? lo : (v > hi ? hi : v);
}

// ---------------------------------------------------------------------------
// Weight pre-conversion: fp32 [OC][nK] -> bf16 [OCpad][nK], zero rows >= OC.
// ---------------------------------------------------------------------------
__global__ void wconv_kernel(__bf16* dst, const float* src, int OC, int nK, int n)
{
    int i = blockIdx.x*blockDim.x + threadIdx.x;
    if (i >= n) return;
    int oc = i / nK;
    dst[i] = (oc < OC) ? f2bf(src[i]) : f2bf(0.f);
}

// ---------------------------------------------------------------------------
// Implicit-GEMM conv (KSxKS, stride 1, pad KS/2) via v_wmma_f32_16x16x32_bf16.
// Block: 32 pixels (one row segment) x NW*16 out-channels. B panel staged in
// LDS K-contiguous per pixel -> fragments are pure b128 loads. Each wave: one
// A fragment feeds two WMMAs (pixel halves).
// ---------------------------------------------------------------------------
template<int IC, int OC, int KS, int NW>
__global__ __launch_bounds__(NW*32)
void conv_wmma_kernel(const float* __restrict__ in, const __bf16* __restrict__ wbf,
                      const float* __restrict__ bias, float* __restrict__ out,
                      int ocTot, int ocBase, int act /*0=none,1=relu*/)
{
    constexpr int KS2 = KS*KS;
    constexpr int nK  = IC*KS2;          // multiple of 32 for all shapes used
    constexpr int nKp = nK + 8;          // row pad: keeps rows 16B aligned
    constexpr int PAD = KS/2;
    constexpr int TW  = 32;              // pixels per block
    __shared__ __align__(16) __bf16 s_B[TW*nKp];

    const int tx   = threadIdx.x;
    const int wave = tx >> 5;
    const int lane = tx & 31;
    const int x0   = blockIdx.x * TW;
    const int by   = blockIdx.y % H_;
    const int b    = blockIdx.y / H_;
    const int oc0  = blockIdx.z * (NW*16) + wave * 16;

    // --- build bf16 im2col panel: s_B[n][k], k = ic*KS2 + ky*KS + kx ---
    for (int idx = tx; idx < TW*nK; idx += NW*32) {
        int k   = idx % nK;
        int n   = idx / nK;
        int ic  = k / KS2;
        int tap = k - ic*KS2;
        int ky  = tap / KS;
        int kx  = tap - ky*KS;
        int y   = by + ky - PAD;
        int x   = x0 + n + kx - PAD;
        float v = 0.f;
        if (y >= 0 && y < H_ && x >= 0 && x < W_)
            v = in[(((size_t)b*IC + ic)*H_ + y)*W_ + x];
        s_B[n*nKp + k] = f2bf(v);
    }
    __syncthreads();

    v8f acc0 = {0.f,0.f,0.f,0.f,0.f,0.f,0.f,0.f};
    v8f acc1 = {0.f,0.f,0.f,0.f,0.f,0.f,0.f,0.f};
    const int mn = lane & 15;
    const int hi = lane >> 4;
    const __bf16* wrow  = wbf + (size_t)(oc0 + mn) * nK;   // padded rows: always valid
    const __bf16* brow0 = s_B + (size_t)mn        * nKp;
    const __bf16* brow1 = s_B + (size_t)(mn + 16) * nKp;

    for (int k0 = 0; k0 < nK; k0 += 32) {
        if (k0 + 128 < nK) __builtin_prefetch(wrow + k0 + 128, 0, 1);
        AFrag A, B0, B1;
        A.h[0]  = *(const v8bf*)(wrow  + k0 + hi*8);        // K k0+hi*8   .. +7
        A.h[1]  = *(const v8bf*)(wrow  + k0 + 16 + hi*8);   // K k0+16+hi*8.. +7
        B0.h[0] = *(const v8bf*)(brow0 + k0 + hi*8);        // K k0+hi*8   .. +7
        B0.h[1] = *(const v8bf*)(brow0 + k0 + 16 + hi*8);   // K k0+16+hi*8.. +7
        B1.h[0] = *(const v8bf*)(brow1 + k0 + hi*8);
        B1.h[1] = *(const v8bf*)(brow1 + k0 + 16 + hi*8);
        acc0 = __builtin_amdgcn_wmma_f32_16x16x32_bf16(
                   false, A.v, false, B0.v, (short)0, acc0, false, false);
        acc1 = __builtin_amdgcn_wmma_f32_16x16x32_bf16(
                   false, A.v, false, B1.v, (short)0, acc1, false, false);
    }

    // D layout: N = lane&15, M = 8*(lane>>4) + vgpr
    const int mrow = hi * 8;
    #pragma unroll
    for (int r = 0; r < 8; ++r) {
        int o = oc0 + mrow + r;
        if (o < OC) {
            float bia = bias[o];
            size_t base = (((size_t)b*ocTot + ocBase + o)*H_ + by)*W_ + x0;
            float v0 = acc0[r] + bia;
            float v1 = acc1[r] + bia;
            if (act == 1) { v0 = v0 > 0.f ? v0 : 0.f; v1 = v1 > 0.f ? v1 : 0.f; }
            out[base + mn]      = v0;
            out[base + 16 + mn] = v1;
        }
    }
}

// ---------------------------------------------------------------------------
// Modulated deformable conv (C=64, 3x3) as WMMA GEMM over K = C*9 = 576.
// Stage 1: bilinear corner indices/weights (mask & validity premultiplied).
// Stage 2: shared bf16 im2col panel via 4-point gathers.
// Stage 3: dual-WMMA K loop identical to conv.
// ---------------------------------------------------------------------------
__global__ __launch_bounds__(128)
void deform_wmma_kernel(const float* __restrict__ in, const float* __restrict__ om,
                        const __bf16* __restrict__ wbf, const float* __restrict__ bias,
                        float* __restrict__ out)
{
    constexpr int nK  = C_*9;    // 576
    constexpr int nKp = nK + 8;
    constexpr int TW  = 32;
    __shared__ int   s_iy0[9][TW], s_iy1[9][TW], s_ix0[9][TW], s_ix1[9][TW];
    __shared__ float s_w00[9][TW], s_w01[9][TW], s_w10[9][TW], s_w11[9][TW];
    __shared__ __align__(16) __bf16 s_B[TW*nKp];

    const int tx   = threadIdx.x;
    const int wave = tx >> 5;
    const int lane = tx & 31;
    const int x0   = blockIdx.x * TW;
    const int by   = blockIdx.y % H_;
    const int b    = blockIdx.y / H_;
    const int oc0  = wave * 16;

    for (int idx = tx; idx < 9*TW; idx += blockDim.x) {
        int t = idx / TW, n = idx % TW;
        int x = x0 + n;
        size_t base = ((size_t)b*27*H_ + by)*W_ + x;   // channel stride HW_
        float dy = om[base + (size_t)(2*t)  *HW_];
        float dx = om[base + (size_t)(2*t+1)*HW_];
        float mk = om[base + (size_t)(18+t) *HW_];     // sigmoid already applied
        float py = (float)by - 1.f + (float)(t/3) + dy;
        float px = (float)x  - 1.f + (float)(t%3) + dx;
        float y0f = floorf(py), x0f = floorf(px);
        float wy = py - y0f, wx = px - x0f;
        bool vy0 = (y0f >= 0.f) && (y0f <= (float)(H_-1));
        bool vy1 = (y0f+1.f >= 0.f) && (y0f+1.f <= (float)(H_-1));
        bool vx0 = (x0f >= 0.f) && (x0f <= (float)(W_-1));
        bool vx1 = (x0f+1.f >= 0.f) && (x0f+1.f <= (float)(W_-1));
        s_iy0[t][n] = iclamp((int)y0f,     0, H_-1);
        s_iy1[t][n] = iclamp((int)y0f + 1, 0, H_-1);
        s_ix0[t][n] = iclamp((int)x0f,     0, W_-1);
        s_ix1[t][n] = iclamp((int)x0f + 1, 0, W_-1);
        s_w00[t][n] = (1.f-wy)*(1.f-wx) * ((vy0 && vx0) ? mk : 0.f);
        s_w01[t][n] = (1.f-wy)*wx       * ((vy0 && vx1) ? mk : 0.f);
        s_w10[t][n] = wy*(1.f-wx)       * ((vy1 && vx0) ? mk : 0.f);
        s_w11[t][n] = wy*wx             * ((vy1 && vx1) ? mk : 0.f);
    }
    __syncthreads();

    for (int idx = tx; idx < TW*nK; idx += 128) {
        int k   = idx % nK;
        int n   = idx / nK;
        int ic  = k / 9;
        int tap = k - ic*9;
        const float* p = in + ((size_t)b*C_ + ic)*(size_t)HW_;
        float v = s_w00[tap][n]*p[s_iy0[tap][n]*W_ + s_ix0[tap][n]]
                + s_w01[tap][n]*p[s_iy0[tap][n]*W_ + s_ix1[tap][n]]
                + s_w10[tap][n]*p[s_iy1[tap][n]*W_ + s_ix0[tap][n]]
                + s_w11[tap][n]*p[s_iy1[tap][n]*W_ + s_ix1[tap][n]];
        s_B[n*nKp + k] = f2bf(v);
    }
    __syncthreads();

    v8f acc0 = {0.f,0.f,0.f,0.f,0.f,0.f,0.f,0.f};
    v8f acc1 = {0.f,0.f,0.f,0.f,0.f,0.f,0.f,0.f};
    const int mn = lane & 15;
    const int hi = lane >> 4;
    const __bf16* wrow  = wbf + (size_t)(oc0 + mn) * nK;
    const __bf16* brow0 = s_B + (size_t)mn        * nKp;
    const __bf16* brow1 = s_B + (size_t)(mn + 16) * nKp;

    for (int k0 = 0; k0 < nK; k0 += 32) {
        AFrag A, B0, B1;
        A.h[0]  = *(const v8bf*)(wrow  + k0 + hi*8);
        A.h[1]  = *(const v8bf*)(wrow  + k0 + 16 + hi*8);
        B0.h[0] = *(const v8bf*)(brow0 + k0 + hi*8);
        B0.h[1] = *(const v8bf*)(brow0 + k0 + 16 + hi*8);
        B1.h[0] = *(const v8bf*)(brow1 + k0 + hi*8);
        B1.h[1] = *(const v8bf*)(brow1 + k0 + 16 + hi*8);
        acc0 = __builtin_amdgcn_wmma_f32_16x16x32_bf16(
                   false, A.v, false, B0.v, (short)0, acc0, false, false);
        acc1 = __builtin_amdgcn_wmma_f32_16x16x32_bf16(
                   false, A.v, false, B1.v, (short)0, acc1, false, false);
    }

    const int mrow = hi * 8;
    #pragma unroll
    for (int r = 0; r < 8; ++r) {
        int o = oc0 + mrow + r;
        float bia = bias[o];
        size_t base = (((size_t)b*C_ + o)*H_ + by)*W_ + x0;
        out[base + mn]      = acc0[r] + bia;
        out[base + 16 + mn] = acc1[r] + bia;
    }
}

// ---------------------------------------------------------------------------
// Small elementwise / reduction helpers
// ---------------------------------------------------------------------------
__global__ void copy_slice_kernel(float* dst, const float* src,
                                  int dstC, int base, int srcC, int n)
{
    int i = blockIdx.x*blockDim.x + threadIdx.x;
    if (i >= n) return;
    int p = i % HW_;
    int r = i / HW_;
    int c = r % srcC;
    int b = r / srcC;
    dst[(((size_t)b*dstC + base + c)*HW_) + p] = src[i];
}

__global__ void mul_slice_kernel(float* dst, const float* a, const float* bsrc,
                                 int dstC, int base, int srcC, int n)
{
    int i = blockIdx.x*blockDim.x + threadIdx.x;
    if (i >= n) return;
    int p = i % HW_;
    int r = i / HW_;
    int c = r % srcC;
    int b = r / srcC;
    dst[(((size_t)b*dstC + base + c)*HW_) + p] = a[i] * bsrc[i];
}

__global__ void leaky_inplace_kernel(float* x, int n)
{
    int i = blockIdx.x*blockDim.x + threadIdx.x;
    if (i >= n) return;
    float v = x[i];
    x[i] = v >= 0.f ? v : 0.1f * v;
}

__global__ void sigmoid_mask_kernel(float* om, int n) // channels 18..26, in place
{
    int i = blockIdx.x*blockDim.x + threadIdx.x;
    if (i >= n) return;                                // n = B*9*HW
    int p  = i % HW_;
    int r  = i / HW_;
    int ch = 18 + (r % 9);
    int b  = r / 9;
    size_t o = (((size_t)b*27 + ch)*HW_) + p;
    om[o] = 1.f / (1.f + __expf(-om[o]));
}

__global__ void mean_kernel(const float* x, float* mean)   // grid = B*RC
{
    __shared__ float red[256];
    const float* p = x + (size_t)blockIdx.x * HW_;
    float s = 0.f;
    for (int i = threadIdx.x; i < HW_; i += blockDim.x) s += p[i];
    red[threadIdx.x] = s;
    __syncthreads();
    for (int st = 128; st > 0; st >>= 1) {
        if ((int)threadIdx.x < st) red[threadIdx.x] += red[threadIdx.x + st];
        __syncthreads();
    }
    if (threadIdx.x == 0) mean[blockIdx.x] = red[0] * (1.f / (float)HW_);
}

__global__ void ca_kernel(const float* mean, const float* wd, const float* bd,
                          const float* wu, const float* bu, float* scale)
{
    __shared__ float hid[12];
    int b = blockIdx.x;
    const float* mb = mean + b*RC_;
    if ((int)threadIdx.x < 12) {
        float s = bd[threadIdx.x];
        for (int i = 0; i < RC_; ++i) s += wd[threadIdx.x*RC_ + i] * mb[i];
        hid[threadIdx.x] = s > 0.f ? s : 0.f;
    }
    __syncthreads();
    if ((int)threadIdx.x < RC_) {
        float s = bu[threadIdx.x];
        for (int j = 0; j < 12; ++j) s += wu[threadIdx.x*12 + j] * hid[j];
        scale[b*RC_ + threadIdx.x] = 1.f / (1.f + __expf(-s));
    }
}

__global__ void axpy_scale_kernel(float* out, const float* x, const float* r,
                                  const float* scale, int n)
{
    int i = blockIdx.x*blockDim.x + threadIdx.x;
    if (i >= n) return;
    out[i] = x[i] + r[i] * scale[i / HW_];   // i/HW == b*RC + ch
}

__global__ void add_kernel(float* out, const float* a, const float* c, int n)
{
    int i = blockIdx.x*blockDim.x + threadIdx.x;
    if (i >= n) return;
    out[i] = a[i] + c[i];
}

// ---------------------------------------------------------------------------
extern "C" void kernel_launch(void* const* d_in, const int* in_sizes, int n_in,
                              void* d_out, int out_size, void* d_ws, size_t ws_size,
                              hipStream_t stream)
{
    (void)in_sizes; (void)n_in; (void)out_size; (void)ws_size;

    const float* curr   = (const float*)d_in[0];
    const float* ngb    = (const float*)d_in[1];
    const float* diff   = (const float*)d_in[2];
    const float* ngbrgb = (const float*)d_in[3];
    const float* intra  = (const float*)d_in[4];

    const float* conv_diff_w = (const float*)d_in[5];
    const float* conv_diff_b = (const float*)d_in[6];
    const float* conv_du_w   = (const float*)d_in[7];
    const float* conv_du_b   = (const float*)d_in[8];
    const float* conv_om_w   = (const float*)d_in[9];
    const float* conv_om_b   = (const float*)d_in[10];
    const float* dcn_w       = (const float*)d_in[11];
    const float* dcn_b       = (const float*)d_in[12];
    const float* out_w       = (const float*)d_in[13];
    const float* out_b       = (const float*)d_in[14];
    const float* r1_c1_w = (const float*)d_in[15];
    const float* r1_c1_b = (const float*)d_in[16];
    const float* r1_c2_w = (const float*)d_in[17];
    const float* r1_c2_b = (const float*)d_in[18];
    const float* r1_cad_w = (const float*)d_in[19];
    const float* r1_cad_b = (const float*)d_in[20];
    const float* r1_cau_w = (const float*)d_in[21];
    const float* r1_cau_b = (const float*)d_in[22];
    const float* r2_c1_w = (const float*)d_in[23];
    const float* r2_c1_b = (const float*)d_in[24];
    const float* r2_c2_w = (const float*)d_in[25];
    const float* r2_c2_b = (const float*)d_in[26];
    const float* r2_cad_w = (const float*)d_in[27];
    const float* r2_cad_b = (const float*)d_in[28];
    const float* r2_cau_w = (const float*)d_in[29];
    const float* r2_cau_b = (const float*)d_in[30];
    const float* tail_w = (const float*)d_in[31];
    const float* tail_b = (const float*)d_in[32];
    const float* re_w   = (const float*)d_in[33];
    const float* re_b   = (const float*)d_in[34];

    char* wsb = (char*)d_ws;
    size_t off = 0;
    auto allocf = [&](size_t nfl)->float* {
        off = (off + 31) & ~(size_t)31;
        float* p = (float*)(wsb + off); off += nfl*4; return p;
    };
    auto allocb = [&](size_t nel)->__bf16* {
        off = (off + 31) & ~(size_t)31;
        __bf16* p = (__bf16*)(wsb + off); off += nel*2; return p;
    };
    const size_t nC  = (size_t)B_ * C_  * HW_;
    const size_t nRC = (size_t)B_ * RC_ * HW_;
    float* t_cat1 = allocf((size_t)B_ * 2*C_ * HW_);   // [curr | conv_diff]
    float* t_d2   = allocf(nC);
    float* t_om   = allocf((size_t)B_ * 27 * HW_);
    float* t_dep  = allocf(nC);
    float* t_rgb  = allocf(nC);
    float* t_cat  = allocf(nRC);                       // [curr | dep2 | rgb*att]
    float* t_a    = allocf(nRC);
    float* t_t1   = allocf(nRC);
    float* t_t2   = allocf(nRC);
    float* t_mean = allocf(B_ * RC_);
    float* t_scale= allocf(B_ * RC_);
    // bf16 weights (zero-padded rows)
    __bf16* wb_diff = allocb((size_t)64 * 576);
    __bf16* wb_du   = allocb((size_t)64 * 128);
    __bf16* wb_om   = allocb((size_t)32 * 576);
    __bf16* wb_dcn  = allocb((size_t)64 * 576);
    __bf16* wb_out  = allocb((size_t)64 * 576);
    __bf16* wb_r1c1 = allocb((size_t)192 * 1728);
    __bf16* wb_r1c2 = allocb((size_t)192 * 1728);
    __bf16* wb_r2c1 = allocb((size_t)192 * 1728);
    __bf16* wb_r2c2 = allocb((size_t)192 * 1728);
    __bf16* wb_tail = allocb((size_t)192 * 1728);
    __bf16* wb_re   = allocb((size_t)64 * 192);

    auto eg = [](size_t n) { return dim3((unsigned)((n + 255) / 256)); };
    auto wcv = [&](__bf16* dst, const float* src, int OC, int nK, int OCpad) {
        int n = OCpad * nK;
        wconv_kernel<<<eg(n), 256, 0, stream>>>(dst, src, OC, nK, n);
    };
    wcv(wb_diff, conv_diff_w, 64, 576, 64);
    wcv(wb_du,   conv_du_w,   64, 128, 64);
    wcv(wb_om,   conv_om_w,   27, 576, 32);
    wcv(wb_dcn,  dcn_w,       64, 576, 64);
    wcv(wb_out,  out_w,       64, 576, 64);
    wcv(wb_r1c1, r1_c1_w,    192, 1728, 192);
    wcv(wb_r1c2, r1_c2_w,    192, 1728, 192);
    wcv(wb_r2c1, r2_c1_w,    192, 1728, 192);
    wcv(wb_r2c2, r2_c2_w,    192, 1728, 192);
    wcv(wb_tail, tail_w,     192, 1728, 192);
    wcv(wb_re,   re_w,        64, 192,  64);

    const dim3 gConv(W_/32, B_*H_, 1);
    const dim3 gConvRC(W_/32, B_*H_, RC_/64);

    // 1) cat1 = [curr, conv_diff(diff_maps)]
    copy_slice_kernel<<<eg(nC), 256, 0, stream>>>(t_cat1, curr, 2*C_, 0, C_, (int)nC);
    conv_wmma_kernel<C_, C_, 3, 4><<<gConv, 128, 0, stream>>>(
        diff, wb_diff, conv_diff_b, t_cat1, 2*C_, C_, 0);
    // 2) d2 = conv_du(cat1)  (1x1, 128->64)
    conv_wmma_kernel<2*C_, C_, 1, 4><<<gConv, 128, 0, stream>>>(
        t_cat1, wb_du, conv_du_b, t_d2, C_, 0, 0);
    // 3) om = conv_om(d2) (64->27), then sigmoid mask channels in place
    conv_wmma_kernel<C_, 27, 3, 2><<<gConv, 64, 0, stream>>>(
        t_d2, wb_om, conv_om_b, t_om, 27, 0, 0);
    sigmoid_mask_kernel<<<eg((size_t)B_*9*HW_), 256, 0, stream>>>(t_om, B_*9*HW_);
    // 4) deformable convs (shared offsets/mask)
    deform_wmma_kernel<<<gConv, 128, 0, stream>>>(ngb,    t_om, wb_dcn, dcn_b, t_dep);
    deform_wmma_kernel<<<gConv, 128, 0, stream>>>(ngbrgb, t_om, wb_dcn, dcn_b, t_rgb);
    // 5) dep2 = conv_out(leaky(dep)) -> cat[:,64:128]
    leaky_inplace_kernel<<<eg(nC), 256, 0, stream>>>(t_dep, (int)nC);
    conv_wmma_kernel<C_, C_, 3, 4><<<gConv, 128, 0, stream>>>(
        t_dep, wb_out, out_b, t_cat, RC_, C_, 0);
    // 6) cat[:,0:64]=curr ; cat[:,128:192]=rgb*intra
    copy_slice_kernel<<<eg(nC), 256, 0, stream>>>(t_cat, curr, RC_, 0, C_, (int)nC);
    mul_slice_kernel<<<eg(nC), 256, 0, stream>>>(t_cat, t_rgb, intra, RC_, 2*C_, C_, (int)nC);

    // 7) RCAB1
    conv_wmma_kernel<RC_, RC_, 3, 4><<<gConvRC, 128, 0, stream>>>(
        t_cat, wb_r1c1, r1_c1_b, t_t1, RC_, 0, 1);
    conv_wmma_kernel<RC_, RC_, 3, 4><<<gConvRC, 128, 0, stream>>>(
        t_t1, wb_r1c2, r1_c2_b, t_t2, RC_, 0, 0);
    mean_kernel<<<B_*RC_, 256, 0, stream>>>(t_t2, t_mean);
    ca_kernel<<<B_, RC_, 0, stream>>>(t_mean, r1_cad_w, r1_cad_b, r1_cau_w, r1_cau_b, t_scale);
    axpy_scale_kernel<<<eg(nRC), 256, 0, stream>>>(t_a, t_cat, t_t2, t_scale, (int)nRC);

    // 8) RCAB2
    conv_wmma_kernel<RC_, RC_, 3, 4><<<gConvRC, 128, 0, stream>>>(
        t_a, wb_r2c1, r2_c1_b, t_t1, RC_, 0, 1);
    conv_wmma_kernel<RC_, RC_, 3, 4><<<gConvRC, 128, 0, stream>>>(
        t_t1, wb_r2c2, r2_c2_b, t_t2, RC_, 0, 0);
    mean_kernel<<<B_*RC_, 256, 0, stream>>>(t_t2, t_mean);
    ca_kernel<<<B_, RC_, 0, stream>>>(t_mean, r2_cad_w, r2_cad_b, r2_cau_w, r2_cau_b, t_scale);
    axpy_scale_kernel<<<eg(nRC), 256, 0, stream>>>(t_t1, t_a, t_t2, t_scale, (int)nRC);

    // 9) tail conv + group residual
    conv_wmma_kernel<RC_, RC_, 3, 4><<<gConvRC, 128, 0, stream>>>(
        t_t1, wb_tail, tail_b, t_t2, RC_, 0, 0);
    add_kernel<<<eg(nRC), 256, 0, stream>>>(t_t2, t_cat, t_t2, (int)nRC);

    // 10) re: 1x1 192->64 into d_out
    conv_wmma_kernel<RC_, C_, 1, 4><<<gConv, 128, 0, stream>>>(
        t_t2, wb_re, re_b, (float*)d_out, C_, 0, 0);
}
